// AttentionBlock_6012954215086
// MI455X (gfx1250) — matrix operand order repaired and checked by reading, not compiled, not run on hardware
//
#include <hip/hip_runtime.h>

// ---------------------------------------------------------------------------
// AttentionBlock for MI455X (gfx1250): all GEMMs via v_wmma_f32_16x16x32_bf16,
// global->LDS staging via async copies (ASYNCcnt) with double buffering.
// B=8, C=512, N=1024 (32x32), GROUPS=8.
// ---------------------------------------------------------------------------

typedef __attribute__((ext_vector_type(16))) __bf16 v16bf;
typedef __attribute__((ext_vector_type(8)))  float  v8f;

union FragBF { v16bf v; uint4 q[2]; };
union FragF  { v8f   v; float f[8]; };

__device__ __forceinline__ unsigned short f32_to_bf16(float f) {
    unsigned int u = __float_as_uint(f);
    unsigned int r = u + 0x7FFFu + ((u >> 16) & 1u);   // round-to-nearest-even
    return (unsigned short)(r >> 16);
}

// Low 32 bits of a generic pointer to __shared__ = LDS offset (flat->LDS rule).
__device__ __forceinline__ unsigned lds_off(const void* p) {
    return (unsigned)(unsigned long long)(uintptr_t)p;
}

// ------------------------- weight conversion -------------------------------
__global__ __launch_bounds__(256)
void cvt_f32_bf16(const float* __restrict__ in, unsigned short* __restrict__ out, int n) {
    int i = blockIdx.x * 256 + threadIdx.x;
    if (i < n) out[i] = f32_to_bf16(in[i]);
}

// ------------------------- GroupNorm ---------------------------------------
__global__ __launch_bounds__(256)
void groupnorm_to_bf16(const float* __restrict__ x,
                       const float* __restrict__ gw,
                       const float* __restrict__ gb,
                       unsigned short* __restrict__ nrmT) {
    const int b = blockIdx.x >> 3;
    const int g = blockIdx.x & 7;
    const float* xg = x + ((size_t)b * 512 + (size_t)g * 64) * 1024;
    const int t = threadIdx.x;

    float s = 0.f, ss = 0.f;
    for (int i = t; i < 65536; i += 256) { float v = xg[i]; s += v; ss += v * v; }

    __shared__ float rs[256], rss[256];
    rs[t] = s; rss[t] = ss;
    __syncthreads();
    for (int off = 128; off > 0; off >>= 1) {
        if (t < off) { rs[t] += rs[t + off]; rss[t] += rss[t + off]; }
        __syncthreads();
    }
    const float mean = rs[0] * (1.f / 65536.f);
    const float var  = rss[0] * (1.f / 65536.f) - mean * mean;
    const float rstd = rsqrtf(var + 1e-5f);

    for (int i = t; i < 65536; i += 256) {
        const int cl = i & 63, n = i >> 6;
        const int c  = g * 64 + cl;
        const float v = (xg[(size_t)cl * 1024 + n] - mean) * rstd * gw[c] + gb[c];
        nrmT[((size_t)b * 1024 + n) * 512 + c] = f32_to_bf16(v);
    }
}

// ------------------------- Softmax rows ------------------------------------
__global__ __launch_bounds__(256)
void softmax_rows(const float* __restrict__ S, unsigned short* __restrict__ P) {
    const size_t row = blockIdx.x;
    const float* sr = S + row * 1024;
    unsigned short* pr = P + row * 1024;
    const int t = threadIdx.x;

    float v[4], mx = -3.4e38f;
#pragma unroll
    for (int j = 0; j < 4; ++j) { v[j] = sr[t + j * 256]; mx = fmaxf(mx, v[j]); }

    __shared__ float red[256];
    red[t] = mx; __syncthreads();
    for (int off = 128; off > 0; off >>= 1) {
        if (t < off) red[t] = fmaxf(red[t], red[t + off]);
        __syncthreads();
    }
    mx = red[0];
    __syncthreads();

    float e[4], sum = 0.f;
#pragma unroll
    for (int j = 0; j < 4; ++j) { e[j] = __expf(v[j] - mx); sum += e[j]; }
    red[t] = sum; __syncthreads();
    for (int off = 128; off > 0; off >>= 1) {
        if (t < off) red[t] += red[t + off];
        __syncthreads();
    }
    const float inv = 1.f / red[0];
#pragma unroll
    for (int j = 0; j < 4; ++j) pr[t + j * 256] = f32_to_bf16(e[j] * inv);
}

// ------------------------- GEMM-NT (WMMA bf16, async double-buffered) -------
// D[m][col] = sum_k A[m][k] * Brows[col][k]   (A,Brows row-major bf16)
// Block tile 128x128, 8 waves of 64x32, K-step 32, LDS double buffer.
// Modes: 0 bf16 out[m*ldd+col](+bias)  1 bf16 out[col*ldd+m](+bias)
//        2 f32  out[m*ldd+col]*scale   4 f32  out[col*ldd+m]+bias+resid
__global__ __launch_bounds__(256)
void gemm_nt_wmma(const unsigned short* __restrict__ A,  int lda, long long sA,
                  const unsigned short* __restrict__ Bw, int ldb, long long sB,
                  void* __restrict__ Dv,                 int ldd, long long sD,
                  int K,
                  const float* __restrict__ bias,
                  const float* __restrict__ resid,       long long sR,
                  float scale, int mode) {
    __shared__ unsigned short As[2][128 * 40];   // 40-elem pitch, 16B-aligned rows
    __shared__ unsigned short Bs[2][128 * 40];
    constexpr unsigned BUFB = 128 * 40 * 2;      // bytes per buffer

    const int tid  = threadIdx.x;
    const int lane = tid & 31;
    const int half = lane >> 4;
    const int r    = lane & 15;
    const int w    = tid >> 5;
    const int wm   = w & 1;
    const int wn   = w >> 1;

    const int bn0 = blockIdx.x * 128;
    const int bm0 = blockIdx.y * 128;
    const int bz  = blockIdx.z;

    const unsigned short* Ab = A  + (size_t)bz * sA;
    const unsigned short* Bb = Bw + (size_t)bz * sB;

    // ---- staging constants: thread -> (row, 16B chunk); 2 rows per operand ----
    const int srow = tid >> 2;      // 0..63
    const int sch  = tid & 3;       // 16B chunk in 64B row
    const unsigned long long gA0 =
        (unsigned long long)(uintptr_t)(Ab + (size_t)(bm0 + srow) * lda + sch * 8);
    const unsigned long long gA1 =
        (unsigned long long)(uintptr_t)(Ab + (size_t)(bm0 + srow + 64) * lda + sch * 8);
    const unsigned long long gB0 =
        (unsigned long long)(uintptr_t)(Bb + (size_t)(bn0 + srow) * ldb + sch * 8);
    const unsigned long long gB1 =
        (unsigned long long)(uintptr_t)(Bb + (size_t)(bn0 + srow + 64) * ldb + sch * 8);
    const unsigned la0 = lds_off(&As[0][srow * 40 + sch * 8]);
    const unsigned la1 = lds_off(&As[0][(srow + 64) * 40 + sch * 8]);
    const unsigned lb0 = lds_off(&Bs[0][srow * 40 + sch * 8]);
    const unsigned lb1 = lds_off(&Bs[0][(srow + 64) * 40 + sch * 8]);

    // 4 async 16B global->LDS copies per thread per K-step (ASYNCcnt += 4/wave)
    auto stage = [&](int buf, int k0) {
        const unsigned long long ko = (unsigned long long)k0 * 2u;
        const unsigned bo = buf ? BUFB : 0u;
        asm volatile("global_load_async_to_lds_b128 %0, %1, off"
                     :: "v"(la0 + bo), "v"(gA0 + ko) : "memory");
        asm volatile("global_load_async_to_lds_b128 %0, %1, off"
                     :: "v"(la1 + bo), "v"(gA1 + ko) : "memory");
        asm volatile("global_load_async_to_lds_b128 %0, %1, off"
                     :: "v"(lb0 + bo), "v"(gB0 + ko) : "memory");
        asm volatile("global_load_async_to_lds_b128 %0, %1, off"
                     :: "v"(lb1 + bo), "v"(gB1 + ko) : "memory");
    };

    FragF acc[4][2];
#pragma unroll
    for (int mt = 0; mt < 4; ++mt)
#pragma unroll
        for (int nt = 0; nt < 2; ++nt) acc[mt][nt].v = v8f{};

    stage(0, 0);                                   // prologue: fill buffer 0
    int cur = 0;
    for (int k0 = 0; k0 < K; k0 += 32) {
        const bool has_next = (k0 + 32) < K;
        __syncthreads();          // buf[cur^1] fully consumed by all waves
        if (has_next) stage(cur ^ 1, k0 + 32);
        if (has_next) asm volatile("s_wait_asynccnt 0x4" ::: "memory");
        else          asm volatile("s_wait_asynccnt 0x0" ::: "memory");
        __syncthreads();          // buf[cur] visible to all waves

        const unsigned short* Asc = As[cur];
        const unsigned short* Bsc = Bs[cur];
        FragBF af[4];
#pragma unroll
        for (int mt = 0; mt < 4; ++mt) {
            const unsigned short* base = &Asc[(wm * 64 + mt * 16 + r) * 40];
            af[mt].q[0] = *reinterpret_cast<const uint4*>(base + half * 8);
            af[mt].q[1] = *reinterpret_cast<const uint4*>(base + 16 + half * 8);
        }
        FragBF bf[2];
#pragma unroll
        for (int nt = 0; nt < 2; ++nt) {
            const unsigned short* base = &Bsc[(wn * 32 + nt * 16 + r) * 40];
            bf[nt].q[0] = *reinterpret_cast<const uint4*>(base + half * 8);
            bf[nt].q[1] = *reinterpret_cast<const uint4*>(base + 16 + half * 8);
        }
#pragma unroll
        for (int mt = 0; mt < 4; ++mt)
#pragma unroll
            for (int nt = 0; nt < 2; ++nt)
                acc[mt][nt].v = __builtin_amdgcn_wmma_f32_16x16x32_bf16(
                    false, af[mt].v, false, bf[nt].v,
                    (short)0, acc[mt][nt].v, false, false);
        cur ^= 1;
    }

    // ------------------------- epilogue -------------------------
    unsigned short* Db = (unsigned short*)Dv + (size_t)bz * sD;
    float*          Df = (float*)Dv          + (size_t)bz * sD;
    const float*    Rb = resid ? resid + (size_t)bz * sR : nullptr;

#pragma unroll
    for (int mt = 0; mt < 4; ++mt) {
#pragma unroll
        for (int nt = 0; nt < 2; ++nt) {
            const int col = bn0 + wn * 32 + nt * 16 + r;
            const int m0  = bm0 + wm * 64 + mt * 16 + half * 8;
            const float bv = bias ? bias[col] : 0.f;
            const FragF& a = acc[mt][nt];
            if (mode == 0) {
                unsigned short* p = Db + (size_t)m0 * ldd + col;
#pragma unroll
                for (int i = 0; i < 8; ++i) p[(size_t)i * ldd] = f32_to_bf16(a.f[i] + bv);
            } else if (mode == 1) {
                unsigned short* p = Db + (size_t)col * ldd + m0;
#pragma unroll
                for (int i = 0; i < 8; ++i) p[i] = f32_to_bf16(a.f[i] + bv);
            } else if (mode == 2) {
                float* p = Df + (size_t)m0 * ldd + col;
#pragma unroll
                for (int i = 0; i < 8; ++i) p[(size_t)i * ldd] = a.f[i] * scale;
            } else {
                float* p = Df + (size_t)col * ldd + m0;
                const float* rp = Rb + (size_t)col * ldd + m0;
#pragma unroll
                for (int i = 0; i < 8; ++i) p[i] = a.f[i] + bv + rp[i];
            }
        }
    }
}

// ---------------------------------------------------------------------------
extern "C" void kernel_launch(void* const* d_in, const int* in_sizes, int n_in,
                              void* d_out, int out_size, void* d_ws, size_t ws_size,
                              hipStream_t stream) {
    (void)in_sizes; (void)n_in; (void)out_size; (void)ws_size;
    const float* x    = (const float*)d_in[0];
    const float* gn_w = (const float*)d_in[1];
    const float* gn_b = (const float*)d_in[2];
    const float* q_w  = (const float*)d_in[3];
    const float* q_b  = (const float*)d_in[4];
    const float* k_w  = (const float*)d_in[5];
    const float* k_b  = (const float*)d_in[6];
    const float* v_w  = (const float*)d_in[7];
    const float* v_b  = (const float*)d_in[8];
    const float* p_w  = (const float*)d_in[9];
    const float* p_b  = (const float*)d_in[10];
    float* out = (float*)d_out;

    const int  Bn = 8, C = 512, N = 1024;
    const long long szAct = (long long)Bn * N * C;
    const long long szS   = (long long)Bn * N * N;
    const int  szW = C * C;

    char* ws = (char*)d_ws;
    size_t off = 0;
    auto alloc = [&](size_t bytes) -> void* {
        void* p = ws + off;
        off = (off + bytes + 255) & ~(size_t)255;
        return p;
    };
    unsigned short* wq   = (unsigned short*)alloc((size_t)szW * 2);
    unsigned short* wk   = (unsigned short*)alloc((size_t)szW * 2);
    unsigned short* wv   = (unsigned short*)alloc((size_t)szW * 2);
    unsigned short* wp   = (unsigned short*)alloc((size_t)szW * 2);
    unsigned short* nrmT = (unsigned short*)alloc((size_t)szAct * 2);
    unsigned short* qT   = (unsigned short*)alloc((size_t)szAct * 2);
    unsigned short* kT   = (unsigned short*)alloc((size_t)szAct * 2);
    unsigned short* vch  = (unsigned short*)alloc((size_t)szAct * 2);
    float*          S    = (float*)alloc((size_t)szS * 4);
    unsigned short* attn = (unsigned short*)alloc((size_t)szS * 2);
    unsigned short* aoT  = (unsigned short*)alloc((size_t)szAct * 2);

    cvt_f32_bf16<<<(szW + 255) / 256, 256, 0, stream>>>(q_w, wq, szW);
    cvt_f32_bf16<<<(szW + 255) / 256, 256, 0, stream>>>(k_w, wk, szW);
    cvt_f32_bf16<<<(szW + 255) / 256, 256, 0, stream>>>(v_w, wv, szW);
    cvt_f32_bf16<<<(szW + 255) / 256, 256, 0, stream>>>(p_w, wp, szW);

    groupnorm_to_bf16<<<64, 256, 0, stream>>>(x, gn_w, gn_b, nrmT);

    const long long sAct = (long long)N * C;
    const long long sSc  = (long long)N * N;

    gemm_nt_wmma<<<dim3(4, 8, 8), 256, 0, stream>>>(nrmT, C, sAct, wq, C, 0,
        qT, C, sAct, C, q_b, nullptr, 0, 1.f, 0);
    gemm_nt_wmma<<<dim3(4, 8, 8), 256, 0, stream>>>(nrmT, C, sAct, wk, C, 0,
        kT, C, sAct, C, k_b, nullptr, 0, 1.f, 0);
    gemm_nt_wmma<<<dim3(4, 8, 8), 256, 0, stream>>>(nrmT, C, sAct, wv, C, 0,
        vch, N, sAct, C, v_b, nullptr, 0, 1.f, 1);

    const float scale = 0.044194173824159216f;      // 512^-0.5
    gemm_nt_wmma<<<dim3(8, 8, 8), 256, 0, stream>>>(qT, C, sAct, kT, C, sAct,
        S, N, sSc, C, nullptr, nullptr, 0, scale, 2);

    softmax_rows<<<Bn * N, 256, 0, stream>>>(S, attn);

    gemm_nt_wmma<<<dim3(4, 8, 8), 256, 0, stream>>>(attn, N, sSc, vch, N, sAct,
        aoT, C, sAct, N, nullptr, nullptr, 0, 1.f, 0);

    gemm_nt_wmma<<<dim3(4, 8, 8), 256, 0, stream>>>(aoT, C, sAct, wp, C, 0,
        out, N, sAct, C, p_b, x, sAct, 1.f, 4);
}
